// Net_Jentzen_42614665511486
// MI455X (gfx1250) — compile-verified, matrix operand order checked by reading, not run on hardware
//
#include <hip/hip_runtime.h>
#include <hip/hip_bf16.h>

// ---------------- problem constants ----------------
namespace {
constexpr int B  = 4096;
constexpr int D  = 256;
constexpr int H  = 266;
constexpr int S  = 50;
constexpr int HP = 288;          // H padded to mult of 32 (used for both GEMM N and K)
constexpr float EPSF = 1e-5f;
// LDS tiling for the GEMM
constexpr int TM = 128;          // block M tile
constexpr int TN = 32;           // block N tile
constexpr int LROW = 80;         // padded LDS row stride in bytes (32 bf16 -> 80B, conflict-free)
constexpr unsigned ABUF0 = 0;
constexpr unsigned ABUF1 = ABUF0 + TM * LROW;          // 10240
constexpr unsigned BBUF0 = ABUF1 + TM * LROW;          // 20480
constexpr unsigned BBUF1 = BBUF0 + TN * LROW;          // 23040
constexpr unsigned SMEM_BYTES = BBUF1 + TN * LROW;     // 25600
}

typedef __attribute__((ext_vector_type(16))) __bf16 bf16x16;
typedef __attribute__((ext_vector_type(8)))  __bf16 bf16x8;
typedef __attribute__((ext_vector_type(8)))  float  f32x8;

__device__ __forceinline__ __bf16 to_bf16(float f) {
  union { float f; unsigned u; } a; a.f = f;
  unsigned u = a.u + 0x7FFFu + ((a.u >> 16) & 1u);   // RNE
  union { unsigned short s; __bf16 b; } r; r.s = (unsigned short)(u >> 16);
  return r.b;
}
__device__ __forceinline__ float bf_to_f(__bf16 b) {
  union { unsigned short s; __bf16 b; } a; a.b = b;
  union { unsigned u; float f; } r; r.u = ((unsigned)a.s) << 16;
  return r.f;
}
// One 16-bit WMMA fragment: elems 0..7 = p[0..7], elems 8..15 = p[16..23]
__device__ __forceinline__ bf16x16 ldfrag(const __bf16* p) {
  bf16x8 lo = *(const bf16x8*)p;
  bf16x8 hi = *(const bf16x8*)(p + 16);
  return __builtin_shufflevector(lo, hi, 0,1,2,3,4,5,6,7,8,9,10,11,12,13,14,15);
}
// CDNA5 async global->LDS copies (ASYNCcnt tracked)
__device__ __forceinline__ void async_b128(unsigned lds_off, const void* gaddr) {
  asm volatile("global_load_async_to_lds_b128 %0, %1, off" :: "v"(lds_off), "v"(gaddr) : "memory");
}
__device__ __forceinline__ void async_b64(unsigned lds_off, const void* gaddr) {
  asm volatile("global_load_async_to_lds_b64 %0, %1, off" :: "v"(lds_off), "v"(gaddr) : "memory");
}

// ---------------- weight convert: f32 (Kact x Nact, row-major) -> bf16 transposed (Npad x Kpad), zero padded
__global__ void k_convert_wt(const float* __restrict__ W, __bf16* __restrict__ Bt,
                             int Kact, int Nact, int Kpad, int Npad,
                             long strideW, long strideBt) {
  const float* Ws = W + (long)blockIdx.y * strideW;
  __bf16* Bts = Bt + (long)blockIdx.y * strideBt;
  int idx = blockIdx.x * blockDim.x + threadIdx.x;
  if (idx >= Npad * Kpad) return;
  int n = idx / Kpad, k = idx - n * Kpad;
  float v = (n < Nact && k < Kact) ? Ws[(long)k * Nact + n] : 0.f;
  Bts[idx] = to_bf16(v);
}

// ---------------- x0 -> bf16 + path[0]
__global__ void k_init_x(const float* __restrict__ x, __bf16* __restrict__ Xbf,
                         float* __restrict__ path0) {
  long i = (long)blockIdx.x * blockDim.x + threadIdx.x;
  if (i >= (long)B * D) return;
  float v = x[i];
  Xbf[i] = to_bf16(v);
  path0[i] = v;
}

// ---------------- LDS-staged WMMA GEMM ----------------
// C[M x Np] = A[M x K](bf16,row) * Bt[Np x K](bf16, pre-transposed) (+bias)
// Block: 256 thr (8 waves) -> 128(M) x 32(N) tile. Per K-chunk(32):
//   A tile 128x32 + B tile 32x32 staged in LDS via async copies, double buffered.
//   Waves arranged 4(M-groups of 32) x 2(N-groups of 16); 2 WMMA per wave per chunk.
__global__ void k_gemm(const __bf16* __restrict__ A, const __bf16* __restrict__ Bt,
                       float* __restrict__ C, const float* __restrict__ bias,
                       int M, int Np, int K, int lda, int ldc, int nbias) {
  __shared__ __align__(16) char smem[SMEM_BYTES];
  const unsigned smb = (unsigned)(uintptr_t)(void*)smem;

  int tid  = threadIdx.x;
  int ntn  = Np >> 5;
  int bm   = blockIdx.x / ntn;
  int bn   = blockIdx.x - bm * ntn;
  int tm0  = bm << 7;                  // block M origin
  int tn0  = bn << 5;                  // block N origin
  int wave = tid >> 5, lane = tid & 31, l = lane & 15, hg = lane >> 4;

  // ---- per-thread async-copy slots ----
  // A tile: 512 x 16B segments; thread covers seg tid and tid+256
  int ar0 = tid >> 2,          ac0 = (tid & 3) << 3;          // row, elem-col
  int ar1 = (tid + 256) >> 2,  ac1 = (tid & 3) << 3;
  // B tile: 256 x 8B segments; thread covers seg tid
  int br = tid >> 3, bc = (tid & 7) << 2;

  const __bf16* Ag0 = A  + (long)(tm0 + ar0) * lda + ac0;
  const __bf16* Ag1 = A  + (long)(tm0 + ar1) * lda + ac1;
  const __bf16* Bg  = Bt + (long)(tn0 + br)  * K   + bc;
  unsigned la0 = (unsigned)(ar0 * LROW) + (unsigned)(ac0 * 2);
  unsigned la1 = (unsigned)(ar1 * LROW) + (unsigned)(ac1 * 2);
  unsigned lb  = (unsigned)(br  * LROW) + (unsigned)(bc  * 2);

  // wave compute coordinates inside the block tile
  int wm = (wave >> 1) << 5;           // 0,32,64,96
  int wn = (wave & 1) << 4;            // 0,16

  int nk = K >> 5;
  // prefetch tile 0 into buffer 0
  async_b128(smb + ABUF0 + la0, Ag0);
  async_b128(smb + ABUF0 + la1, Ag1);
  async_b64 (smb + BBUF0 + lb,  Bg);

  f32x8 acc0 = {}; f32x8 acc1 = {};
  for (int kc = 0; kc < nk; ++kc) {
    if (kc + 1 < nk) {                 // issue next tile into the other buffer
      long go = (long)(kc + 1) << 5;   // element offset
      unsigned ab = ((kc + 1) & 1) ? ABUF1 : ABUF0;
      unsigned bb = ((kc + 1) & 1) ? BBUF1 : BBUF0;
      async_b128(smb + ab + la0, Ag0 + go);
      async_b128(smb + ab + la1, Ag1 + go);
      async_b64 (smb + bb + lb,  Bg  + go);
      asm volatile("s_wait_asynccnt 0x3" ::: "memory");   // tile kc (3 ops) retired
    } else {
      asm volatile("s_wait_asynccnt 0x0" ::: "memory");
    }
    __syncthreads();                   // all waves' pieces of tile kc visible

    unsigned ab = (kc & 1) ? ABUF1 : ABUF0;
    unsigned bb = (kc & 1) ? BBUF1 : BBUF0;
    const char* sm = (const char*)smem;
    const __bf16* pa0 = (const __bf16*)(sm + ab + (wm + l) * LROW) + hg * 8;
    const __bf16* pa1 = (const __bf16*)(sm + ab + (wm + 16 + l) * LROW) + hg * 8;
    const __bf16* pb  = (const __bf16*)(sm + bb + (wn + l) * LROW) + hg * 8;
    bf16x16 fa0 = ldfrag(pa0);
    bf16x16 fa1 = ldfrag(pa1);
    bf16x16 fb  = ldfrag(pb);
    acc0 = __builtin_amdgcn_wmma_f32_16x16x32_bf16(false, fa0, false, fb, (short)0, acc0, false, false);
    acc1 = __builtin_amdgcn_wmma_f32_16x16x32_bf16(false, fa1, false, fb, (short)0, acc1, false, false);
    __syncthreads();                   // everyone done reading tile kc -> safe to overwrite
  }

  int col = tn0 + wn + l;
  float badd = 0.f;
  if (bias && col < nbias) badd = bias[col];
#pragma unroll
  for (int r = 0; r < 8; ++r) {
    int row = tm0 + wm + hg * 8 + r;
    C[(long)row * ldc + col]        = acc0[r] + badd;
    C[(long)(row + 16) * ldc + col] = acc1[r] + badd;
  }
}

// ---------------- BN stats: one block per column; scale = g*rsqrt(var+eps), shift = be - mu*scale
__global__ void k_bn_stats(const float* __restrict__ X, int ldx, int M,
                           const float* __restrict__ g, const float* __restrict__ be,
                           float* __restrict__ scale, float* __restrict__ shift) {
  int col = blockIdx.x, tid = threadIdx.x;
  float s = 0.f, s2 = 0.f;
  for (int r = tid; r < M; r += blockDim.x) {
    float v = X[(long)r * ldx + col];
    s += v; s2 += v * v;
  }
  __shared__ float sh1[256], sh2[256];
  sh1[tid] = s; sh2[tid] = s2; __syncthreads();
  for (int w = 128; w > 0; w >>= 1) {
    if (tid < w) { sh1[tid] += sh1[tid + w]; sh2[tid] += sh2[tid + w]; }
    __syncthreads();
  }
  if (tid == 0) {
    float mu  = sh1[0] / M;
    float var = sh2[0] / M - mu * mu;
    float sc  = g[col] * rsqrtf(var + EPSF);
    scale[col] = sc;
    shift[col] = be[col] - mu * sc;
  }
}

// ---------------- BN apply + ReLU -> padded bf16
__global__ void k_bn_apply(const float* __restrict__ X, int ldx,
                           __bf16* __restrict__ Y, int ldy, int Nact,
                           const float* __restrict__ scale, const float* __restrict__ shift,
                           int M) {
  long i = (long)blockIdx.x * blockDim.x + threadIdx.x;
  if (i >= (long)M * ldy) return;
  int r = (int)(i / ldy), c = (int)(i - (long)r * ldy);
  float o = 0.f;
  if (c < Nact) {
    float v = X[(long)r * ldx + c];
    o = fmaxf(v * scale[c] + shift[c], 0.f);
  }
  Y[i] = to_bf16(o);
}

// ---------------- alpha layer 1 epilogue: relu(P + ab1 + ti*aW1row0) -> padded bf16
__global__ void k_alpha1(const float* __restrict__ X, int ldx,
                         __bf16* __restrict__ Y, int ldy, int Nact,
                         const float* __restrict__ ab1, const float* __restrict__ aW1row0,
                         const float* __restrict__ tg, int s, int M) {
  long i = (long)blockIdx.x * blockDim.x + threadIdx.x;
  if (i >= (long)M * ldy) return;
  int r = (int)(i / ldy), c = (int)(i - (long)r * ldy);
  float o = 0.f;
  if (c < Nact) {
    float ti = tg[s];
    o = fmaxf(X[(long)r * ldx + c] + ab1[c] + ti * aW1row0[c], 0.f);
  }
  Y[i] = to_bf16(o);
}

// ---------------- v-net final column: per-row dot(H2, vW3) + vb3
__global__ void k_vdot(const __bf16* __restrict__ H2, int ld,
                       const float* __restrict__ vW3, const float* __restrict__ vb3,
                       float* __restrict__ out, int M, int Kact) {
  int r = blockIdx.x * blockDim.x + threadIdx.x;
  if (r >= M) return;
  const __bf16* hr = H2 + (long)r * ld;
  float acc = vb3[0];
  for (int k = 0; k < Kact; ++k) acc += bf_to_f(hr[k]) * vW3[k];
  out[r] = acc;
}

// ---------------- BN(+relu) on the single v column, single block
__global__ void k_vbn(const float* __restrict__ tmp, float* __restrict__ v0,
                      const float* __restrict__ g, const float* __restrict__ be, int M) {
  int tid = threadIdx.x;
  float s = 0.f, s2 = 0.f;
  for (int r = tid; r < M; r += 256) { float v = tmp[r]; s += v; s2 += v * v; }
  __shared__ float sh1[256], sh2[256];
  __shared__ float sc, sf;
  sh1[tid] = s; sh2[tid] = s2; __syncthreads();
  for (int w = 128; w > 0; w >>= 1) {
    if (tid < w) { sh1[tid] += sh1[tid + w]; sh2[tid] += sh2[tid + w]; }
    __syncthreads();
  }
  if (tid == 0) {
    float mu  = sh1[0] / M;
    float var = sh2[0] / M - mu * mu;
    sc = g[0] * rsqrtf(var + EPSF);
    sf = be[0] - mu * sc;
  }
  __syncthreads();
  for (int r = tid; r < M; r += 256) v0[r] = fmaxf(tmp[r] * sc + sf, 0.f);
}

// ---------------- SDE step finish: one block per batch row, 256 threads = D
__global__ void k_step(float* __restrict__ path, float* __restrict__ v,
                       float* __restrict__ xf, __bf16* __restrict__ Xbf,
                       const float* __restrict__ G, const float* __restrict__ AL,
                       const float* __restrict__ xi, const float* __restrict__ law,
                       const float* __restrict__ tg, int s, int lastStep) {
  int row = blockIdx.x;
  int d = threadIdx.x;
  const float* xc = path + ((long)s * B + row) * D;
  float* xn = path + ((long)(s + 1) * B + row) * D;
  float h  = tg[s + 1] - tg[s];
  float sq = sqrtf(h);                       // SIGMA = 1
  float xcv = xc[d];
  float al  = AL[(long)row * D + d];
  float gr  = G[(long)row * D + d];
  float xiv = xi[(long)row * D + d];
  float dl  = xcv - law[(long)s * D + d];    // KAPPA = 1

  __shared__ float shf[256], shg[256];
  shf[d] = dl * dl + al * al;
  shg[d] = gr * xiv;
  __syncthreads();
  for (int w = 128; w > 0; w >>= 1) {
    if (d < w) { shf[d] += shf[d + w]; shg[d] += shg[d + w]; }
    __syncthreads();
  }
  if (d == 0) {
    float f   = 0.5f * shf[0];
    float ito = sq * shg[0];
    v[row] = v[row] - f * h + ito;
  }
  float xnv = xcv + al * h + sq * xiv;
  xn[d] = xnv;
  Xbf[(long)row * D + d] = to_bf16(xnv);
  if (lastStep) xf[(long)row * D + d] = xnv;
}

// =============================================================================
extern "C" void kernel_launch(void* const* d_in, const int* in_sizes, int n_in,
                              void* d_out, int out_size, void* d_ws, size_t ws_size,
                              hipStream_t stream) {
  (void)in_sizes; (void)n_in; (void)out_size; (void)ws_size;
  const float* x    = (const float*)d_in[0];
  const float* W1   = (const float*)d_in[1];
  const float* g1   = (const float*)d_in[3];
  const float* be1  = (const float*)d_in[4];
  const float* W2   = (const float*)d_in[5];
  const float* g2   = (const float*)d_in[7];
  const float* be2  = (const float*)d_in[8];
  const float* W3   = (const float*)d_in[9];
  const float* b3   = (const float*)d_in[10];
  const float* vW1  = (const float*)d_in[11];
  const float* vg1  = (const float*)d_in[13];
  const float* vbe1 = (const float*)d_in[14];
  const float* vW2  = (const float*)d_in[15];
  const float* vg2  = (const float*)d_in[17];
  const float* vbe2 = (const float*)d_in[18];
  const float* vW3  = (const float*)d_in[19];
  const float* vb3  = (const float*)d_in[20];
  const float* vg3  = (const float*)d_in[21];
  const float* vbe3 = (const float*)d_in[22];
  const float* aW1  = (const float*)d_in[23];
  const float* ab1  = (const float*)d_in[24];
  const float* aW2  = (const float*)d_in[25];
  const float* ab2  = (const float*)d_in[26];
  const float* law  = (const float*)d_in[27];
  const float* tg   = (const float*)d_in[28];
  const float* xi   = (const float*)d_in[29];

  // outputs: v (B), xf (B*D), path (T*B*D)
  float* vout = (float*)d_out;
  float* xfo  = vout + B;
  float* path = xfo + (long)B * D;

  // ---- workspace layout ----
  char* wsb = (char*)d_ws;
  size_t off = 0;
  auto alloc = [&](size_t bytes) { size_t o = off; off = (off + bytes + 255) & ~(size_t)255; return o; };
  __bf16* W1t  = (__bf16*)(wsb + alloc((size_t)S * HP * D  * 2));
  __bf16* W2t  = (__bf16*)(wsb + alloc((size_t)S * HP * HP * 2));
  __bf16* W3t  = (__bf16*)(wsb + alloc((size_t)S * D  * HP * 2));
  __bf16* aW1t = (__bf16*)(wsb + alloc((size_t)HP * D  * 2));
  __bf16* aW2t = (__bf16*)(wsb + alloc((size_t)D  * HP * 2));
  __bf16* vW1t = (__bf16*)(wsb + alloc((size_t)HP * D  * 2));
  __bf16* vW2t = (__bf16*)(wsb + alloc((size_t)HP * HP * 2));
  __bf16* Xbf  = (__bf16*)(wsb + alloc((size_t)B * D  * 2));
  __bf16* H1bf = (__bf16*)(wsb + alloc((size_t)B * HP * 2));
  __bf16* H2bf = (__bf16*)(wsb + alloc((size_t)B * HP * 2));
  __bf16* A1bf = (__bf16*)(wsb + alloc((size_t)B * HP * 2));
  float*  S0   = (float*) (wsb + alloc((size_t)B * HP * 4));
  float*  Gm   = (float*) (wsb + alloc((size_t)B * D * 4));
  float*  AL   = (float*) (wsb + alloc((size_t)B * D * 4));
  float*  scB  = (float*) (wsb + alloc((size_t)HP * 4));
  float*  shB  = (float*) (wsb + alloc((size_t)HP * 4));
  float*  vtmp = (float*) (wsb + alloc((size_t)B * 4));

  auto cvt = [&](const float* W, __bf16* Bt, int Kact, int Nact, int Kpad, int Npad,
                 long sw, long sb, int steps) {
    dim3 grid((Npad * Kpad + 255) / 256, steps);
    k_convert_wt<<<grid, 256, 0, stream>>>(W, Bt, Kact, Nact, Kpad, Npad, sw, sb);
  };
  auto gemm = [&](const __bf16* A, const __bf16* Bt, float* C, const float* bias,
                  int Np, int K, int lda, int ldc, int nbias) {
    int blocks = (B / TM) * (Np / TN);
    k_gemm<<<blocks, 256, 0, stream>>>(A, Bt, C, bias, B, Np, K, lda, ldc, nbias);
  };
  auto bnpair = [&](const float* g, const float* be, __bf16* Y) {
    k_bn_stats<<<H, 256, 0, stream>>>(S0, HP, B, g, be, scB, shB);
    k_bn_apply<<<(B * HP) / 256, 256, 0, stream>>>(S0, HP, Y, HP, H, scB, shB, B);
  };

  // ---- weight conversion (transpose + pad + bf16) ----
  cvt(W1, W1t, D, H, D,  HP, (long)D * H, (long)HP * D,  S);
  cvt(W2, W2t, H, H, HP, HP, (long)H * H, (long)HP * HP, S);
  cvt(W3, W3t, H, D, HP, D,  (long)H * D, (long)D * HP,  S);
  cvt(aW1 + H, aW1t, D, H, D,  HP, 0, 0, 1);   // rows 1..256 of aW1 (row 0 = time feature)
  cvt(aW2,     aW2t, H, D, HP, D,  0, 0, 1);
  cvt(vW1,     vW1t, D, H, D,  HP, 0, 0, 1);
  cvt(vW2,     vW2t, H, H, HP, HP, 0, 0, 1);

  // ---- x0 -> bf16 + path[0] ----
  k_init_x<<<(B * D) / 256, 256, 0, stream>>>(x, Xbf, path);

  // ---- value network on x0 (v0) ----
  gemm(Xbf, vW1t, S0, nullptr, HP, D, D, HP, 0);
  bnpair(vg1, vbe1, H1bf);
  gemm(H1bf, vW2t, S0, nullptr, HP, HP, HP, HP, 0);
  bnpair(vg2, vbe2, H2bf);
  k_vdot<<<B / 256, 256, 0, stream>>>(H2bf, HP, vW3, vb3, vtmp, B, H);
  k_vbn<<<1, 256, 0, stream>>>(vtmp, vout, vg3, vbe3, B);

  // ---- 50-step scan ----
  for (int s = 0; s < S; ++s) {
    // h1 = relu(BN(x @ W1_s))
    gemm(Xbf, W1t + (size_t)s * HP * D, S0, nullptr, HP, D, D, HP, 0);
    bnpair(g1 + (size_t)s * H, be1 + (size_t)s * H, H1bf);
    // h2 = relu(BN(h1 @ W2_s))
    gemm(H1bf, W2t + (size_t)s * HP * HP, S0, nullptr, HP, HP, HP, HP, 0);
    bnpair(g2 + (size_t)s * H, be2 + (size_t)s * H, H2bf);
    // grad = h2 @ W3_s + b3_s
    gemm(H2bf, W3t + (size_t)s * D * HP, Gm, b3 + (size_t)s * D, D, HP, HP, D, D);
    // alpha = relu(x @ aW1' + ab1 + t*aW1[0]) @ aW2 + ab2
    gemm(Xbf, aW1t, S0, nullptr, HP, D, D, HP, 0);
    k_alpha1<<<(B * HP) / 256, 256, 0, stream>>>(S0, HP, A1bf, HP, H, ab1, aW1, tg, s, B);
    gemm(A1bf, aW2t, AL, ab2, D, HP, HP, D, D);
    // SDE update: path[s+1], v, Xbf (and xf on last step)
    k_step<<<B, D, 0, stream>>>(path, vout, xfo, Xbf, Gm, AL, xi + (size_t)s * B * D,
                                law, tg, s, s == S - 1 ? 1 : 0);
  }
}
